// GCNTemporalRNN_7713761263791
// MI455X (gfx1250) — matrix operand order repaired and checked by reading, not compiled, not run on hardware
//
#include <hip/hip_runtime.h>
#include <hip/hip_fp16.h>

#define T_STEPS 12
#define HT 64
#define HID 128
#define OUT_F 12

typedef __attribute__((ext_vector_type(16))) _Float16 v16h;
typedef __attribute__((ext_vector_type(8)))  _Float16 v8h;
typedef __attribute__((ext_vector_type(4)))  _Float16 v4h;
typedef __attribute__((ext_vector_type(8)))  float    v8f;
typedef __attribute__((ext_vector_type(4)))  float    v4f;

__device__ __forceinline__ float sigmoidf_(float v) {
    return 1.0f / (1.0f + __expf(-v));
}

// ---------------------------------------------------------------------------
// GRU over T steps: gh = h @ w_hh^T via v_wmma_f32_16x16x32_f16.
// One wave per 16-node tile; 8 waves / block.
// h kept in LDS twice: f32 master (recurrence update) + f16 shadow (A frags).
// ---------------------------------------------------------------------------
__global__ void __launch_bounds__(256)
gru_kernel(const float* __restrict__ x,      // [N, T]
           const float* __restrict__ w_ih,   // [192, 1]
           const float* __restrict__ w_hh,   // [192, 64] row-major
           const float* __restrict__ b_ih,   // [192]
           const float* __restrict__ b_hh,   // [192]
           _Float16* __restrict__ hOut,      // [N, 64] f16
           int N)
{
    __shared__ alignas(32) _Float16 sWhh[3 * HT * HT];   // 24 KB, row-major [192][64]
    __shared__ float sWih[3 * HT];
    __shared__ float sBih[3 * HT];
    __shared__ float sBhh[3 * HT];
    __shared__ float sH[8][16 * HT];                     // 32 KB, per-wave h (f32 master)
    __shared__ alignas(16) _Float16 sHh[8][16 * HT];     // 16 KB, f16 shadow for A frags
    __shared__ float sX[8][16];

    const int tid = threadIdx.x;
    for (int i = tid; i < 3 * HT * HT; i += 256) sWhh[i] = (_Float16)w_hh[i];
    for (int i = tid; i < 3 * HT; i += 256) {
        sWih[i] = w_ih[i];
        sBih[i] = b_ih[i];
        sBhh[i] = b_hh[i];
    }
    const int wave = tid >> 5;
    const int lane = tid & 31;
    for (int i = lane; i < 16 * HT; i += 32) {
        sH[wave][i]  = 0.0f;
        sHh[wave][i] = (_Float16)0.0f;
    }
    __syncthreads();

    const int nodeBase = (blockIdx.x * 8 + wave) * 16;
    const int r  = lane & 15;
    const int hi = lane >> 4;
    const _Float16* hph = &sHh[wave][r * HT];  // lane's matrix row (A layout: M = lane&15)

    for (int t = 0; t < T_STEPS; ++t) {
        if (lane < 16) {
            int node = nodeBase + lane;
            sX[wave][lane] = (node < N) ? x[(size_t)node * T_STEPS + t] : 0.0f;
        }
        // A fragments for K=[0,32) and [32,64): elements 0..15 -> K b0..b0+15,
        // b0 = hi*16.  4x ds_load_b128 total.
        const int b0 = hi * 16;
        v8h p0 = *(const v8h*)(hph + b0);
        v8h p1 = *(const v8h*)(hph + b0 + 8);
        v8h p2 = *(const v8h*)(hph + 32 + b0);
        v8h p3 = *(const v8h*)(hph + 32 + b0 + 8);
        v16h a0 = __builtin_shufflevector(p0, p1, 0,1,2,3,4,5,6,7,8,9,10,11,12,13,14,15);
        v16h a1 = __builtin_shufflevector(p2, p3, 0,1,2,3,4,5,6,7,8,9,10,11,12,13,14,15);

        // 12 column tiles of gh (192 = 3 gates * 64)
        v8f acc[12];
        #pragma unroll
        for (int c = 0; c < 12; ++c) {
            v8f z = {};
            // B(k,n) = w_hh[(16c+n)*64 + k]; lane n=r needs K kb + hi*16 .. +15 contiguous
            const v16h* bp0 = (const v16h*)&sWhh[(c * 16 + r) * HT + hi * 16];
            const v16h* bp1 = (const v16h*)&sWhh[(c * 16 + r) * HT + 32 + hi * 16];
            z = __builtin_amdgcn_wmma_f32_16x16x32_f16(false, a0, false, *bp0, (short)0, z, false, false);
            z = __builtin_amdgcn_wmma_f32_16x16x32_f16(false, a1, false, *bp1, (short)0, z, false, false);
            acc[c] = z;
        }
        // Gate math directly on D layout: reg -> row (reg + hi*8), col = cc*16 + r
        #pragma unroll
        for (int cc = 0; cc < 4; ++cc) {
            const int j = cc * 16 + r;
            const float wir = sWih[j],       bir  = sBih[j],       bhr = sBhh[j];
            const float wiz = sWih[64 + j],  biz  = sBih[64 + j],  bhz = sBhh[64 + j];
            const float win = sWih[128 + j], bin_ = sBih[128 + j], bhn = sBhh[128 + j];
            #pragma unroll
            for (int reg = 0; reg < 8; ++reg) {
                const int rowIT = reg + hi * 8;
                const float xv = sX[wave][rowIT];
                const float rg = sigmoidf_(xv * wir + bir + acc[cc][reg] + bhr);
                const float zg = sigmoidf_(xv * wiz + biz + acc[cc + 4][reg] + bhz);
                const float ng = tanhf(xv * win + bin_ + rg * (acc[cc + 8][reg] + bhn));
                const float ho = sH[wave][rowIT * HT + j];
                const float hn = (1.0f - zg) * ng + zg * ho;
                sH[wave][rowIT * HT + j]  = hn;
                sHh[wave][rowIT * HT + j] = (_Float16)hn;
            }
        }
        __syncthreads();
    }
    for (int i = lane; i < 16 * HT; i += 32) {
        int node = nodeBase + (i >> 6);
        if (node < N) hOut[(size_t)node * HT + (i & 63)] = sHh[wave][i];
    }
}

// ---------------------------------------------------------------------------
// Generic WMMA GEMM: C[N x fvalid] = A(f16 [N,K]) @ W(f32 [K,fvalid]) (+bias)
// W staged transposed [F][K] f16 in LDS; one wave per 16-row tile.
// ---------------------------------------------------------------------------
template <int K, int F>
__global__ void __launch_bounds__(256)
gemm_kernel(const _Float16* __restrict__ A, const float* __restrict__ W,
            const float* __restrict__ bias, float* __restrict__ C,
            int N, int ldc, int fvalid)
{
    __shared__ alignas(32) _Float16 sW[F * K];   // transposed [F][K]
    const int tid = threadIdx.x;
    for (int i = tid; i < K * F; i += 256) {
        int k = i / F, f = i % F;
        float w = (f < fvalid) ? W[k * fvalid + f] : 0.0f;
        sW[f * K + k] = (_Float16)w;
    }
    __syncthreads();

    const int wave = tid >> 5, lane = tid & 31;
    const int r = lane & 15, hi = lane >> 4;
    const int rowBase = (blockIdx.x * 8 + wave) * 16;
    const int row = rowBase + r;

    constexpr int KS = K / 32;
    v16h afrag[KS];
    #pragma unroll
    for (int ks = 0; ks < KS; ++ks) {
        v16h a = {};
        if (row < N) {
            const _Float16* ap = A + (size_t)row * K + ks * 32 + hi * 16;
            v8h lo = *(const v8h*)ap;          // K +0..7  (relative)
            v8h hv = *(const v8h*)(ap + 8);    // K +8..15
            a = __builtin_shufflevector(lo, hv, 0,1,2,3,4,5,6,7,8,9,10,11,12,13,14,15);
        }
        afrag[ks] = a;
    }
    #pragma unroll
    for (int c = 0; c < F / 16; ++c) {
        v8f acc = {};
        #pragma unroll
        for (int ks = 0; ks < KS; ++ks) {
            const v16h* bp = (const v16h*)&sW[(c * 16 + r) * K + ks * 32 + hi * 16];
            acc = __builtin_amdgcn_wmma_f32_16x16x32_f16(false, afrag[ks], false, *bp, (short)0, acc, false, false);
        }
        #pragma unroll
        for (int reg = 0; reg < 8; ++reg) {
            const int orow = rowBase + reg + hi * 8;
            const int ocol = c * 16 + r;
            if (orow < N && ocol < fvalid) {
                float v = acc[reg];
                if (bias) v += bias[ocol];
                C[(size_t)orow * ldc + ocol] = v;
            }
        }
    }
}

// ---------------------------------------------------------------------------
// Graph-norm + scatter helpers
// ---------------------------------------------------------------------------
__global__ void fill_zero(float* p, size_t n) {
    size_t i = blockIdx.x * (size_t)blockDim.x + threadIdx.x;
    if (i < n) p[i] = 0.0f;
}

__global__ void deg_kernel(const int* __restrict__ ei, const float* __restrict__ ew,
                           float* __restrict__ deg, int E)
{
    int e = blockIdx.x * blockDim.x + threadIdx.x;
    if (e < E) atomicAdd(&deg[ei[(size_t)E + e]], ew[e]);   // segment_sum over col
}

__global__ void dinv_kernel(const float* __restrict__ deg, float* __restrict__ dinv, int N)
{
    int i = blockIdx.x * blockDim.x + threadIdx.x;
    if (i < N) {
        float d = deg[i] + 1.0f;                            // + self-loop weight
        dinv[i] = d > 0.0f ? rsqrtf(d) : 0.0f;
    }
}

// one WAVE per edge (8 edges/block); lane handles 4 features via float4 load.
// agg (25.6 MB) is L2-resident on MI455X (192 MB L2) -> atomics resolve at L2.
__global__ void __launch_bounds__(256)
scatter_kernel(const int* __restrict__ ei, const float* __restrict__ ew,
               const float* __restrict__ dinv, const float* __restrict__ xw,
               float* __restrict__ agg, int E, int F)
{
    const int wave = threadIdx.x >> 5;
    const int lane = threadIdx.x & 31;
    const int e = blockIdx.x * 8 + wave;
    if (e >= E) return;
    const int rs = ei[e];                    // wave-uniform -> scalar loads
    const int cs = ei[(size_t)E + e];
    const float nrm = dinv[rs] * ew[e] * dinv[cs];
    const v4f v = *(const v4f*)(xw + (size_t)rs * F + lane * 4);   // global_load_b128
    float* dst = agg + (size_t)cs * F + lane * 4;
    #pragma unroll
    for (int j = 0; j < 4; ++j) atomicAdd(dst + j, v[j] * nrm);
}

// self-loop term + bias + relu + f16 convert (float4 in, 4x f16 out)
__global__ void finish_kernel(const float* __restrict__ agg, const float* __restrict__ xw,
                              const float* __restrict__ dinv, const float* __restrict__ bias,
                              _Float16* __restrict__ hOut, int N, int F)
{
    size_t q = blockIdx.x * (size_t)blockDim.x + threadIdx.x;   // quad index
    if (q < (size_t)N * F / 4) {
        size_t base = q * 4;
        int i  = (int)(base / F);
        int f0 = (int)(base % F);
        float di = dinv[i];
        float s = di * di;
        v4f a = *(const v4f*)(agg + base);
        v4f w = *(const v4f*)(xw + base);
        v4h o;
        #pragma unroll
        for (int j = 0; j < 4; ++j)
            o[j] = (_Float16)fmaxf(a[j] + s * w[j] + bias[f0 + j], 0.0f);
        *(v4h*)(hOut + base) = o;
    }
}

// ---------------------------------------------------------------------------
extern "C" void kernel_launch(void* const* d_in, const int* in_sizes, int n_in,
                              void* d_out, int out_size, void* d_ws, size_t ws_size,
                              hipStream_t stream)
{
    const float* x    = (const float*)d_in[0];
    const int*   ei   = (const int*)d_in[1];
    const float* ew   = (const float*)d_in[2];
    const float* w_ih = (const float*)d_in[3];
    const float* w_hh = (const float*)d_in[4];
    const float* b_ih = (const float*)d_in[5];
    const float* b_hh = (const float*)d_in[6];
    const float* W1   = (const float*)d_in[7];
    const float* b1   = (const float*)d_in[8];
    const float* W2   = (const float*)d_in[9];
    const float* b2   = (const float*)d_in[10];
    const float* Wfc  = (const float*)d_in[11];
    const float* bfc  = (const float*)d_in[12];
    float* out = (float*)d_out;

    const int N = in_sizes[0] / T_STEPS;   // 50000
    const int E = in_sizes[2];             // 600000

    char* ws = (char*)d_ws;
    auto align256 = [](size_t v) { return (v + 255) & ~(size_t)255; };
    size_t off = 0;
    _Float16* hF16  = (_Float16*)(ws + off); off = align256(off + (size_t)N * HT * 2);
    float*    xw    = (float*)(ws + off);    off = align256(off + (size_t)N * HID * 4);
    float*    agg   = (float*)(ws + off);    off = align256(off + (size_t)N * HID * 4);
    _Float16* h1F16 = (_Float16*)(ws + off); off = align256(off + (size_t)N * HID * 2);
    float*    deg   = (float*)(ws + off);    off = align256(off + (size_t)N * 4);
    float*    dinv  = (float*)(ws + off);    off = align256(off + (size_t)N * 4);
    (void)ws_size; (void)n_in; (void)out_size;

    const int rowTiles   = (N + 15) / 16;        // 3125
    const int gemmBlocks = (rowTiles + 7) / 8;   // 391 blocks of 8 waves
    const int nfQuads    = (int)(((size_t)N * HID / 4 + 255) / 256);
    const int scatBlocks = (E + 7) / 8;

    // temporal GRU -> h_last (f16)
    gru_kernel<<<gemmBlocks, 256, 0, stream>>>(x, w_ih, w_hh, b_ih, b_hh, hF16, N);

    // sym-norm degree
    fill_zero<<<(N + 255) / 256, 256, 0, stream>>>(deg, (size_t)N);
    deg_kernel<<<(E + 255) / 256, 256, 0, stream>>>(ei, ew, deg, E);
    dinv_kernel<<<(N + 255) / 256, 256, 0, stream>>>(deg, dinv, N);

    // GCN layer 1
    gemm_kernel<HT, HID><<<gemmBlocks, 256, 0, stream>>>(hF16, W1, nullptr, xw, N, HID, HID);
    fill_zero<<<(int)(((size_t)N * HID + 255) / 256), 256, 0, stream>>>(agg, (size_t)N * HID);
    scatter_kernel<<<scatBlocks, 256, 0, stream>>>(ei, ew, dinv, xw, agg, E, HID);
    finish_kernel<<<nfQuads, 256, 0, stream>>>(agg, xw, dinv, b1, h1F16, N, HID);

    // GCN layer 2 (reuses xw/agg; h1F16 overwritten in place after gemm consumed it)
    gemm_kernel<HID, HID><<<gemmBlocks, 256, 0, stream>>>(h1F16, W2, nullptr, xw, N, HID, HID);
    fill_zero<<<(int)(((size_t)N * HID + 255) / 256), 256, 0, stream>>>(agg, (size_t)N * HID);
    scatter_kernel<<<scatBlocks, 256, 0, stream>>>(ei, ew, dinv, xw, agg, E, HID);
    finish_kernel<<<nfQuads, 256, 0, stream>>>(agg, xw, dinv, b2, h1F16, N, HID);

    // linear head (OUT=12 padded to a 16-wide WMMA tile)
    gemm_kernel<HID, 16><<<gemmBlocks, 256, 0, stream>>>(h1F16, Wfc, bfc, out, N, OUT_F, OUT_F);
}
